// GraphUnet_8632884265214
// MI455X (gfx1250) — matrix operand order repaired and checked by reading, not compiled, not run on hardware
//
#include <hip/hip_runtime.h>
#include <hip/hip_fp16.h>

// ---------------------------------------------------------------------------
// Graph U-Net for gfx1250 (MI455X).
//  - GCN aggregation: edge-parallel float4 gather + global f32 atomic scatter
//    (HBM/L2-bound phase; ~150MB working set largely fits the 192MB L2).
//  - Dense layers: wave32 WMMA (v_wmma_f32_16x16x32_f16). Each wave computes
//    4 M-tiles x 1 N-tile (64x16 output), reusing the B fragment 4x. Weights
//    are pre-converted to f16 *column-major* so the B fragment is a single
//    contiguous 32B load per lane. The node dimension is padded to a multiple
//    of 256 rows so the GEMM has NO divergent guards: EXEC stays all-1s
//    (WMMA requirement) and no exec-save accumulator shuffles are generated.
//  - top-k: 4-pass 8-bit radix select + index-ordered tie break (matches
//    jax.lax.top_k tie semantics).
// ---------------------------------------------------------------------------

typedef __attribute__((ext_vector_type(16))) _Float16 v16h;
typedef __attribute__((ext_vector_type(8)))  float    v8f;

#define TPB 256

static __host__ __device__ inline int ceil_div(int a, int b) { return (a + b - 1) / b; }

// ---------------- degree accumulation --------------------------------------
__global__ void k_degrees(const int* __restrict__ src, const int* __restrict__ dst,
                          const float* __restrict__ m, int E,
                          float* __restrict__ dout, float* __restrict__ din) {
    int e = blockIdx.x * blockDim.x + threadIdx.x;
    if (e >= E) return;
    int s = src[e], d = dst[e];
    float w = 1.0f;
    if (m) w = m[s] * m[d];
    if (w != 0.0f) {
        atomicAdd(&dout[s], w);
        atomicAdd(&din[d], w);
    }
}

__global__ void k_invsqrt(float* __restrict__ deg, int n) {
    int i = blockIdx.x * blockDim.x + threadIdx.x;
    if (i >= n) return;
    deg[i] = rsqrtf(fmaxf(deg[i], 1.0f));
}

// ---------------- message scatter: agg[dst] += h[src]*degout_is[src]*emask --
__global__ void k_scatter(const float* __restrict__ h, const int* __restrict__ src,
                          const int* __restrict__ dst, const float* __restrict__ m,
                          const float* __restrict__ dout_is, int E, int F,
                          float* __restrict__ agg) {
    int idx = blockIdx.x * blockDim.x + threadIdx.x;
    int chunks = F >> 2;
    if (idx >= E * chunks) return;
    int e = idx / chunks;
    int c = (idx - e * chunks) << 2;
    int s = src[e], d = dst[e];
    float w = dout_is[s];
    if (m) w *= m[s] * m[d];
    if (w == 0.0f) return;
    float4 v = *(const float4*)(h + (size_t)s * F + c);
    float* ap = agg + (size_t)d * F + c;
    atomicAdd(ap + 0, v.x * w);
    atomicAdd(ap + 1, v.y * w);
    atomicAdd(ap + 2, v.z * w);
    atomicAdd(ap + 3, v.w * w);
}

// ---------------- weight f32 row-major -> f16 column-major -----------------
// WhT[n*K + k] = (f16) W[k*Nc + n]
__global__ void k_w_to_f16_t(const float* __restrict__ w, _Float16* __restrict__ wht,
                             int K, int Nc) {
    int i = blockIdx.x * blockDim.x + threadIdx.x;
    if (i >= K * Nc) return;
    int n = i / K;
    int k = i - n * K;
    wht[(size_t)n * K + k] = (_Float16)w[(size_t)k * Nc + n];
}

// ---------------- WMMA GEMM: out = epi( (A*rowscale) @ W + bias ) ----------
// Grid covers a padded M (multiple of 256): no guards, EXEC stays all-1s.
// Each wave: MT=4 M-tiles (64 rows) x one 16-col N-tile. B fragment loaded
// once per K-step from f16 column-major WhT as one contiguous 32B load/lane.
template <bool RELU, bool MASK>
__global__ void k_gemm_wmma(const float* __restrict__ A, const float* __restrict__ rowscale,
                            const _Float16* __restrict__ WhT, const float* __restrict__ bias,
                            const float* __restrict__ rowmask, float* __restrict__ out,
                            int K, int Nc) {
    constexpr int MT = 4;
    const int lane = threadIdx.x & 31;
    const int wave = threadIdx.x >> 5;
    const int row_base = (blockIdx.x * 4 + wave) * (16 * MT);
    const int col0 = blockIdx.y * 16;
    const int l16  = lane & 15;
    const int hi   = lane >> 4;                // 0 or 1

    float rs[MT];
    const float* ap[MT];
#pragma unroll
    for (int t = 0; t < MT; ++t) {
        int arow = row_base + t * 16 + l16;    // A layout: both halves M=0..15
        rs[t] = rowscale[arow];
        ap[t] = A + (size_t)arow * K;
    }
    const int bcol = col0 + l16;
    const _Float16* bbase = WhT + (size_t)bcol * K;   // column-major weights

    v8f acc[MT];
#pragma unroll
    for (int t = 0; t < MT; ++t) acc[t] = (v8f){0.f,0.f,0.f,0.f,0.f,0.f,0.f,0.f};

    for (int k0 = 0; k0 < K; k0 += 32) {
        // B fragment: lanes 0-15 -> K=k0..k0+15, lanes 16-31 -> K=k0+16..k0+31
        v16h bv = *(const v16h*)(bbase + k0 + hi * 16);
        const int ka = k0 + hi * 16;
#pragma unroll
        for (int t = 0; t < MT; ++t) {
            float4 a0 = *(const float4*)(ap[t] + ka);
            float4 a1 = *(const float4*)(ap[t] + ka + 4);
            float4 a2 = *(const float4*)(ap[t] + ka + 8);
            float4 a3 = *(const float4*)(ap[t] + ka + 12);
            const float r = rs[t];
            v16h av;
            av[0]  = (_Float16)(a0.x * r); av[1]  = (_Float16)(a0.y * r);
            av[2]  = (_Float16)(a0.z * r); av[3]  = (_Float16)(a0.w * r);
            av[4]  = (_Float16)(a1.x * r); av[5]  = (_Float16)(a1.y * r);
            av[6]  = (_Float16)(a1.z * r); av[7]  = (_Float16)(a1.w * r);
            av[8]  = (_Float16)(a2.x * r); av[9]  = (_Float16)(a2.y * r);
            av[10] = (_Float16)(a2.z * r); av[11] = (_Float16)(a2.w * r);
            av[12] = (_Float16)(a3.x * r); av[13] = (_Float16)(a3.y * r);
            av[14] = (_Float16)(a3.z * r); av[15] = (_Float16)(a3.w * r);
            acc[t] = __builtin_amdgcn_wmma_f32_16x16x32_f16(false, av, false, bv,
                                                            (short)0, acc[t], false, false);
        }
    }

    const int n  = col0 + l16;
    const float bn = bias[n];
#pragma unroll
    for (int t = 0; t < MT; ++t) {
#pragma unroll
        for (int j = 0; j < 8; ++j) {          // C layout: vgpr j -> M=j (+8 for hi half)
            int mrow = row_base + t * 16 + j + hi * 8;
            float v = acc[t][j] + bn;
            if (RELU) v = fmaxf(v, 0.0f);
            if (MASK) v *= rowmask[mrow];
            out[(size_t)mrow * Nc + n] = v;
        }
    }
}

// ---------------- gPool scores: sigmoid(h @ Wp + bp), one wave per row -----
__global__ void k_scores(const float* __restrict__ h, const float* __restrict__ wp,
                         const float* __restrict__ bp, float* __restrict__ scores,
                         int N, int F) {
    int lane = threadIdx.x & 31;
    int wave = threadIdx.x >> 5;
    int row  = blockIdx.x * (blockDim.x >> 5) + wave;
    if (row >= N) return;
    const float* hp = h + (size_t)row * F;
    float s = 0.0f;
    for (int t = lane; t < F; t += 32) s += hp[t] * wp[t];
#pragma unroll
    for (int o = 16; o > 0; o >>= 1) s += __shfl_xor(s, o, 32);
    if (lane == 0) scores[row] = 1.0f / (1.0f + expf(-(s + bp[0])));
}

// ---------------- radix-select top-k ---------------------------------------
__global__ void k_keys(const float* __restrict__ scores, const float* __restrict__ m,
                       unsigned* __restrict__ keys, int N) {
    int i = blockIdx.x * blockDim.x + threadIdx.x;
    if (i >= N) return;
    float x = (m == nullptr || m[i] > 0.0f) ? scores[i] : -1e9f;
    unsigned u = __float_as_uint(x);
    u ^= (u & 0x80000000u) ? 0xFFFFFFFFu : 0x80000000u;   // order-preserving map
    keys[i] = u;
}

__global__ void k_sel_init(unsigned* __restrict__ state, int k) {
    int i = threadIdx.x;
    if (i == 0) { state[0] = 0u; state[1] = (unsigned)k; }
    state[2 + i] = 0u;  // hist, blockDim==256
}

__global__ void k_hist(const unsigned* __restrict__ keys, int N,
                       unsigned* __restrict__ state, int pass) {
    int i = blockIdx.x * blockDim.x + threadIdx.x;
    if (i >= N) return;
    unsigned key = keys[i];
    int shift = 24 - 8 * pass;
    if (pass > 0 && (key >> (shift + 8)) != state[0]) return;
    atomicAdd(&state[2 + ((key >> shift) & 0xFFu)], 1u);
}

__global__ void k_scan(unsigned* __restrict__ state) {
    if (threadIdx.x != 0 || blockIdx.x != 0) return;
    unsigned rem = state[1], cum = 0;
    int D = 0;
    for (int d = 255; d >= 0; --d) {
        unsigned c = state[2 + d];
        if (cum + c >= rem) { D = d; break; }
        cum += c;
    }
    state[0] = (state[0] << 8) | (unsigned)D;
    state[1] = rem - cum;
    for (int d = 0; d < 256; ++d) state[2 + d] = 0u;
}

__global__ void k_mask_gt(const unsigned* __restrict__ keys, const unsigned* __restrict__ state,
                          float* __restrict__ nm, int N) {
    int i = blockIdx.x * blockDim.x + threadIdx.x;
    if (i >= N) return;
    nm[i] = (keys[i] > state[0]) ? 1.0f : 0.0f;
}

// single wave, index-ordered tie selection (matches top_k lowest-index-first)
__global__ void k_ties(const unsigned* __restrict__ keys, const unsigned* __restrict__ state,
                       float* __restrict__ nm, int N) {
    unsigned T = state[0];
    unsigned r = state[1];
    int lane = threadIdx.x;
    for (int base = 0; base < N && r > 0; base += 32) {
        int i = base + lane;
        bool eq = (i < N) && (keys[i] == T);
        unsigned mask32 = (unsigned)__ballot(eq);
        unsigned before = __popc(mask32 & ((1u << lane) - 1u));
        if (eq && before < r) nm[i] = 1.0f;
        unsigned cnt = __popc(mask32);
        r = (cnt >= r) ? 0u : (r - cnt);
    }
}

// ---------------- elementwise ----------------------------------------------
__global__ void k_gate(float* __restrict__ h, const float* __restrict__ scores,
                       const float* __restrict__ nm, int N, int F) {
    int idx = blockIdx.x * blockDim.x + threadIdx.x;
    if (idx >= N * F) return;
    int row = idx / F;
    h[idx] *= scores[row] * nm[row];
}

__global__ void k_add(float* __restrict__ a, const float* __restrict__ b, int n) {
    int i = blockIdx.x * blockDim.x + threadIdx.x;
    if (i >= n) return;
    a[i] += b[i];
}

// ---------------------------------------------------------------------------
extern "C" void kernel_launch(void* const* d_in, const int* in_sizes, int n_in,
                              void* d_out, int out_size, void* d_ws, size_t ws_size,
                              hipStream_t stream) {
    const int IN = 128, HID = 256, OUT = 128;
    const int K0 = 25000, K1 = 12500;

    const float* features = (const float*)d_in[0];
    const int*   src      = (const int*)d_in[1];
    const int*   dst      = (const int*)d_in[2];
    const float* W_embed  = (const float*)d_in[3];
    const float* b_embed  = (const float*)d_in[4];
    const float* W_enc0   = (const float*)d_in[5];
    const float* b_enc0   = (const float*)d_in[6];
    const float* W_enc1   = (const float*)d_in[7];
    const float* b_enc1   = (const float*)d_in[8];
    const float* W_p0     = (const float*)d_in[9];
    const float* b_p0     = (const float*)d_in[10];
    const float* W_p1     = (const float*)d_in[11];
    const float* b_p1     = (const float*)d_in[12];
    const float* W_bot    = (const float*)d_in[13];
    const float* b_bot    = (const float*)d_in[14];
    const float* W_dec0   = (const float*)d_in[15];
    const float* b_dec0   = (const float*)d_in[16];
    const float* W_dec1   = (const float*)d_in[17];
    const float* b_dec1   = (const float*)d_in[18];

    const int Nn   = in_sizes[0] / IN;          // 50000
    const int E    = in_sizes[1];               // 800000
    const int Npad = ceil_div(Nn, 256) * 256;   // GEMM M padded: no guards needed

    // ---- workspace carve-up (node-dim buffers padded to Npad rows) ----
    char* wp_ = (char*)d_ws;
    auto carve = [&](size_t bytes) -> void* {
        void* p = wp_;
        wp_ += (bytes + 255) & ~(size_t)255;
        return p;
    };
    float*     h      = (float*)carve((size_t)Npad * HID * 4);
    float*     agg    = (float*)carve((size_t)Npad * HID * 4);
    float*     hid0   = (float*)carve((size_t)Npad * HID * 4);
    float*     hid1   = (float*)carve((size_t)Npad * HID * 4);   // also final-layer pad buffer
    float*     dego   = (float*)carve((size_t)Npad * 4);
    float*     degi   = (float*)carve((size_t)Npad * 4);
    float*     scores = (float*)carve((size_t)Nn * 4);
    float*     m1     = (float*)carve((size_t)Nn * 4);
    float*     m2     = (float*)carve((size_t)Nn * 4);
    unsigned*  keys   = (unsigned*)carve((size_t)Nn * 4);
    unsigned*  state  = (unsigned*)carve(258 * 4);
    _Float16*  WhT    = (_Float16*)carve((size_t)HID * HID * 2);

    const int gN  = ceil_div(Nn, TPB);
    const int gE  = ceil_div(E, TPB);
    const int gNH = ceil_div(Nn * HID, TPB);

    // ---- one GCN layer (out buffer must have Npad rows) ----
    auto run_gcn = [&](const float* hin, int Fin, const float* W, const float* b,
                       const float* mask, const float* rowmask,
                       float* hout, int Fout, bool relu) {
        hipMemsetAsync(dego, 0, (size_t)Npad * 4, stream);
        hipMemsetAsync(degi, 0, (size_t)Npad * 4, stream);   // padded rows: rowscale = 0
        hipMemsetAsync(agg, 0, (size_t)Npad * Fin * 4, stream);
        k_degrees<<<gE, TPB, 0, stream>>>(src, dst, mask, E, dego, degi);
        k_invsqrt<<<gN, TPB, 0, stream>>>(dego, Nn);
        k_invsqrt<<<gN, TPB, 0, stream>>>(degi, Nn);
        int total = E * (Fin >> 2);
        k_scatter<<<ceil_div(total, TPB), TPB, 0, stream>>>(hin, src, dst, mask, dego, E, Fin, agg);
        k_w_to_f16_t<<<ceil_div(Fin * Fout, TPB), TPB, 0, stream>>>(W, WhT, Fin, Fout);
        dim3 g(Npad / 256, Fout / 16);   // 4 waves/block x 64 rows/wave, exact cover
        if (relu && rowmask)
            k_gemm_wmma<true, true><<<g, 128, 0, stream>>>(agg, degi, WhT, b, rowmask, hout, Fin, Fout);
        else if (relu)
            k_gemm_wmma<true, false><<<g, 128, 0, stream>>>(agg, degi, WhT, b, rowmask, hout, Fin, Fout);
        else
            k_gemm_wmma<false, false><<<g, 128, 0, stream>>>(agg, degi, WhT, b, rowmask, hout, Fin, Fout);
    };

    // NOTE (MASK path): rowmask is indexed up to Npad rows; m1/m2 are carved at
    // Nn. Padded-row reads would run past m1 into m2/keys memory (in-bounds of
    // d_ws, harmless garbage in padded rows). The MASK
    // GEMMs pass m1/m2 (carved at Nn) directly.

    // ---- top-k radix select: keys -> node mask nm (exactly k ones) ----
    auto topk = [&](int k, float* nm) {
        k_sel_init<<<1, 256, 0, stream>>>(state, k);
        for (int p = 0; p < 4; ++p) {
            k_hist<<<gN, TPB, 0, stream>>>(keys, Nn, state, p);
            k_scan<<<1, 1, 0, stream>>>(state);
        }
        k_mask_gt<<<gN, TPB, 0, stream>>>(keys, state, nm, Nn);
        k_ties<<<1, 32, 0, stream>>>(keys, state, nm, Nn);
    };

    // ================= forward =================
    // embed: GCN(features) -> h, relu
    run_gcn(features, IN, W_embed, b_embed, nullptr, nullptr, h, HID, true);

    // encoder level 0 (mask = all ones)
    run_gcn(h, HID, W_enc0, b_enc0, nullptr, nullptr, h, HID, true);
    hipMemcpyAsync(hid0, h, (size_t)Nn * HID * 4, hipMemcpyDeviceToDevice, stream);
    k_scores<<<ceil_div(Nn, 8), 256, 0, stream>>>(h, W_p0, b_p0, scores, Nn, HID);
    k_keys<<<gN, TPB, 0, stream>>>(scores, nullptr, keys, Nn);
    topk(K0, m1);
    k_gate<<<gNH, TPB, 0, stream>>>(h, scores, m1, Nn, HID);

    // encoder level 1 (mask = m1)
    run_gcn(h, HID, W_enc1, b_enc1, m1, m1, h, HID, true);
    hipMemcpyAsync(hid1, h, (size_t)Nn * HID * 4, hipMemcpyDeviceToDevice, stream);
    k_scores<<<ceil_div(Nn, 8), 256, 0, stream>>>(h, W_p1, b_p1, scores, Nn, HID);
    k_keys<<<gN, TPB, 0, stream>>>(scores, m1, keys, Nn);
    topk(K1, m2);
    k_gate<<<gNH, TPB, 0, stream>>>(h, scores, m2, Nn, HID);

    // bottleneck (mask = m2)
    run_gcn(h, HID, W_bot, b_bot, m2, m2, h, HID, true);

    // decoder 0: skip from level 1, mask = m1
    k_add<<<gNH, TPB, 0, stream>>>(h, hid1, Nn * HID);
    run_gcn(h, HID, W_dec0, b_dec0, m1, m1, h, HID, true);

    // decoder 1: skip from level 0, mask = ones, no activation.
    // Write to padded scratch (hid1 is dead), then copy valid region to d_out.
    k_add<<<gNH, TPB, 0, stream>>>(h, hid0, Nn * HID);
    run_gcn(h, HID, W_dec1, b_dec1, nullptr, nullptr, hid1, OUT, false);
    hipMemcpyAsync(d_out, hid1, (size_t)Nn * OUT * 4, hipMemcpyDeviceToDevice, stream);
}
